// RNN_69114613730079
// MI455X (gfx1250) — matrix operand order repaired and checked by reading, not compiled, not run on hardware
//
#include <hip/hip_runtime.h>
#include <hip/hip_bf16.h>
#include <math.h>
#include <stdint.h>

#define B_     256
#define R_     49
#define FEA_   2048
#define E_     512
#define H_     1024
#define V_     10000
#define VP_    10240     // V padded to multiple of 128 for tiling
#define T_     20
#define STEPS_ 19
#define G4_    4096      // 4*H
#define KX_    1536      // E + H (fused gates K)

// LDS A-tile geometry: 128 rows x 32 bf16, rows padded 64B -> 80B (bank-conflict-free)
#define ROWB_   80
#define TILEB_  (128 * ROWB_)   // 10240 B per buffer

typedef __bf16 bf16;
typedef __attribute__((ext_vector_type(16))) __bf16 v16bf;
typedef __attribute__((ext_vector_type(8)))  __bf16 v8bf;
typedef __attribute__((ext_vector_type(8)))  float  v8f;

union AFrag {
    v16bf v;
    struct { v8bf lo; v8bf hi; } p;
};

// ---------------- prep kernels ----------------

// fea[b][f] = mean_r fea_maps[b][r][f]  (bf16 result, fp32 accumulate)
__global__ __launch_bounds__(256) void mean_kernel(const float* __restrict__ fm,
                                                   bf16* __restrict__ fea) {
    int idx = blockIdx.x * 256 + threadIdx.x;
    if (idx >= B_ * FEA_) return;
    int b = idx / FEA_, f = idx % FEA_;
    const float* p = fm + (size_t)b * R_ * FEA_ + f;
    float s = 0.f;
    for (int r = 0; r < R_; ++r) s += p[(size_t)r * FEA_];
    fea[idx] = (bf16)(s * (1.f / 49.f));
}

// scale[row] = g[row] / ||v[row,:]||   (one block per row)
__global__ __launch_bounds__(256) void rownorm_kernel(const float* __restrict__ v,
                                                      const float* __restrict__ g,
                                                      float* __restrict__ scale, int K) {
    __shared__ float red[256];
    int row = blockIdx.x;
    const float* p = v + (size_t)row * K;
    float s = 0.f;
    for (int i = threadIdx.x; i < K; i += 256) { float x = p[i]; s += x * x; }
    red[threadIdx.x] = s;
    __syncthreads();
    for (int off = 128; off; off >>= 1) {
        if (threadIdx.x < off) red[threadIdx.x] += red[threadIdx.x + off];
        __syncthreads();
    }
    if (threadIdx.x == 0) scale[row] = g[row] * rsqrtf(red[0]);
}

// dst[k*Npad + n] = bf16(src[n*K + k] * scale[n]), zero-pad n >= N
__global__ __launch_bounds__(256) void tcast_kernel(const float* __restrict__ src,
                                                    const float* __restrict__ scale,
                                                    bf16* __restrict__ dst,
                                                    int N, int K, int Npad) {
    int idx = blockIdx.x * 256 + threadIdx.x;
    if (idx >= K * Npad) return;
    int n = idx % Npad;
    int k = idx / Npad;
    dst[idx] = (n < N) ? (bf16)(src[(size_t)n * K + k] * scale[n]) : (bf16)0.0f;
}

// WgT[k][n] = k < E ? W_ih[n][k] : W_hh[n][k-E]   (K-major, bf16)
__global__ __launch_bounds__(256) void concat_cast_kernel(const float* __restrict__ Wih,
                                                          const float* __restrict__ Whh,
                                                          bf16* __restrict__ dst) {
    int idx = blockIdx.x * 256 + threadIdx.x;
    if (idx >= KX_ * G4_) return;
    int n = idx % G4_;
    int k = idx / G4_;
    float v = (k < E_) ? Wih[(size_t)n * E_ + k] : Whh[(size_t)n * H_ + (k - E_)];
    dst[idx] = (bf16)v;
}

__global__ __launch_bounds__(256) void bias_sum_kernel(const float* __restrict__ a,
                                                       const float* __restrict__ b,
                                                       float* __restrict__ d) {
    int idx = blockIdx.x * 256 + threadIdx.x;
    if (idx < G4_) d[idx] = a[idx] + b[idx];
}

// xh[b][0:E] = bf16(embed_w[caption[b][t]][:])
__global__ __launch_bounds__(256) void pack_x_kernel(const int* __restrict__ cap,
                                                     const float* __restrict__ emb,
                                                     bf16* __restrict__ xh, int t) {
    int idx = blockIdx.x * 256 + threadIdx.x;
    if (idx >= B_ * E_) return;
    int b = idx / E_, e = idx % E_;
    int tok = cap[b * T_ + t];
    xh[(size_t)b * KX_ + e] = (bf16)emb[(size_t)tok * E_ + e];
}

// xh[b][E + j] = bf16(h[b][j])
__global__ __launch_bounds__(256) void cast_h_kernel(const float* __restrict__ h,
                                                     bf16* __restrict__ xh) {
    int idx = blockIdx.x * 256 + threadIdx.x;
    if (idx >= B_ * H_) return;
    int b = idx / H_, j = idx % H_;
    xh[(size_t)b * KX_ + E_ + j] = (bf16)h[idx];
}

__device__ __forceinline__ float sigf(float x) { return 1.f / (1.f + __expf(-x)); }

// LSTM cell pointwise; also refreshes bf16 h inside xh for the next GEMMs
__global__ __launch_bounds__(256) void lstm_cell_kernel(const float* __restrict__ gates,
                                                        float* __restrict__ h,
                                                        float* __restrict__ c,
                                                        bf16* __restrict__ xh) {
    int idx = blockIdx.x * 256 + threadIdx.x;
    if (idx >= B_ * H_) return;
    int b = idx / H_, j = idx % H_;
    const float* g = gates + (size_t)b * G4_;
    float ig = sigf(g[j]);
    float fg = sigf(g[H_ + j]);
    float gg = tanhf(g[2 * H_ + j]);
    float og = sigf(g[3 * H_ + j]);
    float cn = fg * c[idx] + ig * gg;
    float hn = og * tanhf(cn);
    c[idx] = cn;
    h[idx] = hn;
    xh[(size_t)b * KX_ + E_ + j] = (bf16)hn;
}

// ---------------- WMMA bf16 GEMM with async-LDS A staging ----------------
// C[M,N] = A[M,K] * Bt[K,N] + bias[N]
// A: row-major bf16, lda elems. Bt: K-major bf16 (row k holds N outputs), ldb elems.
// Block tile 128M x 128N, 8 waves (4M x 2N), wave tile 32M x 64N.
// A tile (128x32 bf16 per K-step) is staged into LDS via
// global_load_async_to_lds_b128 (ASYNCcnt) and double-buffered; rows are
// padded to 80B so the 16 ds_load_b128 start banks (20L mod 64) are all
// distinct -> conflict-free. B is streamed from L2.
// Grid: (N/128, M/128). Requires M %128==0, K %32==0, N %128==0.
__global__ __launch_bounds__(256) void gemm_bf16_wmma(const bf16* __restrict__ A, int lda,
                                                      const bf16* __restrict__ Bt, int ldb,
                                                      const float* __restrict__ bias,
                                                      float* __restrict__ C, int ldc,
                                                      int K, int Nreal) {
    __shared__ char smem[2 * TILEB_];

    const int tid  = threadIdx.x;
    const int lane = tid & 31;
    const int wave = tid >> 5;
    const int waveM = wave >> 1;          // 0..3
    const int waveN = wave & 1;           // 0..1
    const int m0 = waveM * 32;
    const int n0 = blockIdx.x * 128 + waveN * 64;

    // ---- async A-tile prefetch geometry: 512 16B chunks, 2 per thread ----
    const uint32_t ldsbase = (uint32_t)(uintptr_t)&smem[0];   // LDS byte offset (ISA 10.2)
    const uint64_t Ablk = (uint64_t)(uintptr_t)A + (uint64_t)blockIdx.y * 128u * (uint64_t)(lda * 2);
    const int rowp  = tid >> 2;                // 0..63 (chunk rows; +64 for 2nd chunk)
    const int colpB = (tid & 3) << 4;          // 0/16/32/48 bytes within row

    // A fragment LDS addressing (row stride ROWB_)
    const uint32_t aOff = (uint32_t)(m0 + (lane & 15)) * (uint32_t)ROWB_
                        + (uint32_t)((lane >> 4) << 4);

    v8f acc0 = {}, acc1 = {}, acc2 = {}, acc3 = {};
    v8f acc4 = {}, acc5 = {}, acc6 = {}, acc7 = {};

    const int nk = K >> 5;

#define PREFETCH_A(kk, buf)                                                              \
    do {                                                                                 \
        uint64_t g0 = Ablk + (uint64_t)rowp * (uint64_t)(lda * 2) + (uint64_t)((kk)*2)   \
                      + (uint64_t)colpB;                                                 \
        uint64_t g1 = g0 + (uint64_t)64u * (uint64_t)(lda * 2);                          \
        uint32_t l0 = ldsbase + (uint32_t)((buf) * TILEB_)                               \
                      + (uint32_t)(rowp * ROWB_) + (uint32_t)colpB;                      \
        uint32_t l1 = l0 + (uint32_t)(64 * ROWB_);                                       \
        asm volatile("global_load_async_to_lds_b128 %0, %1, off"                         \
                     :: "v"(l0), "v"(g0) : "memory");                                    \
        asm volatile("global_load_async_to_lds_b128 %0, %1, off"                         \
                     :: "v"(l1), "v"(g1) : "memory");                                    \
    } while (0)

    PREFETCH_A(0, 0);
    asm volatile("s_wait_asynccnt 0x0" ::: "memory");
    __syncthreads();

    for (int i = 0; i < nk; ++i) {
        const int k = i << 5;
        if (i + 1 < nk) PREFETCH_A(k + 32, (i + 1) & 1);

        const char* lbuf = smem + (i & 1) * TILEB_;
        AFrag a0, a1;
        const v8bf* lp0 = (const v8bf*)(lbuf + aOff);
        const v8bf* lp1 = (const v8bf*)(lbuf + aOff + 16 * ROWB_);
        a0.p.lo = lp0[0];  a0.p.hi = lp0[2];   // K 0..7 / 16..23 (lanes 0-15), 8..15 / 24..31 (16-31)
        a1.p.lo = lp1[0];  a1.p.hi = lp1[2];

        // B fragments (32x16 bf16): lane L holds column N=L%16, K 0..7 / 16..23 (lanes 0-15), 8..15 / 24..31 (16-31)
        const bf16* Bp = Bt + (size_t)(k + ((lane >> 4) << 3)) * ldb + n0 + (lane & 15);
        v16bf b0, b1, b2, b3;
#pragma unroll
        for (int s = 0; s < 16; ++s) {
            const bf16* q = Bp + (size_t)((s & 7) + ((s >> 3) << 4)) * ldb;
            b0[s] = q[0]; b1[s] = q[16]; b2[s] = q[32]; b3[s] = q[48];
        }

        acc0 = __builtin_amdgcn_wmma_f32_16x16x32_bf16(false, a0.v, false, b0, (short)0, acc0, false, false);
        acc1 = __builtin_amdgcn_wmma_f32_16x16x32_bf16(false, a0.v, false, b1, (short)0, acc1, false, false);
        acc2 = __builtin_amdgcn_wmma_f32_16x16x32_bf16(false, a0.v, false, b2, (short)0, acc2, false, false);
        acc3 = __builtin_amdgcn_wmma_f32_16x16x32_bf16(false, a0.v, false, b3, (short)0, acc3, false, false);
        acc4 = __builtin_amdgcn_wmma_f32_16x16x32_bf16(false, a1.v, false, b0, (short)0, acc4, false, false);
        acc5 = __builtin_amdgcn_wmma_f32_16x16x32_bf16(false, a1.v, false, b1, (short)0, acc5, false, false);
        acc6 = __builtin_amdgcn_wmma_f32_16x16x32_bf16(false, a1.v, false, b2, (short)0, acc6, false, false);
        acc7 = __builtin_amdgcn_wmma_f32_16x16x32_bf16(false, a1.v, false, b3, (short)0, acc7, false, false);

        asm volatile("s_wait_asynccnt 0x0" ::: "memory");
        __syncthreads();
    }
#undef PREFETCH_A

    // C/D layout: vgpr r, lanes 0-15 -> (M=r, N=lane), lanes 16-31 -> (M=8+r, N=lane-16)
    const int mrow = blockIdx.y * 128 + m0 + ((lane >> 4) << 3);
    const int ncol = n0 + (lane & 15);
    v8f accA[8] = {acc0, acc1, acc2, acc3, acc4, acc5, acc6, acc7};
#pragma unroll
    for (int half = 0; half < 2; ++half) {
#pragma unroll
        for (int s = 0; s < 4; ++s) {
            int n = ncol + s * 16;
            if (n < Nreal) {
                float bb = bias ? bias[n] : 0.f;
#pragma unroll
                for (int r = 0; r < 8; ++r)
                    C[(size_t)(mrow + half * 16 + r) * ldc + n] = accA[half * 4 + s][r] + bb;
            }
        }
    }
}

// ---------------- launcher ----------------

extern "C" void kernel_launch(void* const* d_in, const int* in_sizes, int n_in,
                              void* d_out, int out_size, void* d_ws, size_t ws_size,
                              hipStream_t stream) {
    (void)in_sizes; (void)n_in; (void)out_size; (void)ws_size;
    const float* fea_maps = (const float*)d_in[0];
    const int*   caption  = (const int*)d_in[1];
    const float* embed_w  = (const float*)d_in[2];
    const float* W_ih     = (const float*)d_in[3];
    const float* W_hh     = (const float*)d_in[4];
    const float* b_ih     = (const float*)d_in[5];
    const float* b_hh     = (const float*)d_in[6];
    const float* fc_v     = (const float*)d_in[7];
    const float* fc_g     = (const float*)d_in[8];
    const float* fc_b     = (const float*)d_in[9];
    const float* ih_v     = (const float*)d_in[10];
    const float* ih_g     = (const float*)d_in[11];
    const float* ih_b     = (const float*)d_in[12];
    const float* ic_v     = (const float*)d_in[13];
    const float* ic_g     = (const float*)d_in[14];
    const float* ic_b     = (const float*)d_in[15];
    float* out = (float*)d_out;

    char* ws = (char*)d_ws;
    size_t off = 0;
    auto nb = [&](size_t bytes) { size_t o = off; off += (bytes + 255) & ~(size_t)255; return o; };

    bf16*  fea_bf = (bf16*)(ws + nb((size_t)B_ * FEA_ * 2));
    bf16*  WihT   = (bf16*)(ws + nb((size_t)FEA_ * H_ * 2));
    bf16*  WicT   = (bf16*)(ws + nb((size_t)FEA_ * H_ * 2));
    bf16*  WfcT   = (bf16*)(ws + nb((size_t)H_ * VP_ * 2));
    bf16*  WgT    = (bf16*)(ws + nb((size_t)KX_ * G4_ * 2));
    bf16*  xh     = (bf16*)(ws + nb((size_t)B_ * KX_ * 2));
    float* hbuf   = (float*)(ws + nb((size_t)B_ * H_ * 4));
    float* cbuf   = (float*)(ws + nb((size_t)B_ * H_ * 4));
    float* gates  = (float*)(ws + nb((size_t)B_ * G4_ * 4));
    float* ihsc   = (float*)(ws + nb((size_t)H_ * 4));
    float* icsc   = (float*)(ws + nb((size_t)H_ * 4));
    float* fcsc   = (float*)(ws + nb((size_t)V_ * 4));
    float* bsum   = (float*)(ws + nb((size_t)G4_ * 4));

    // ---- prep ----
    mean_kernel<<<(B_ * FEA_) / 256, 256, 0, stream>>>(fea_maps, fea_bf);

    rownorm_kernel<<<H_, 256, 0, stream>>>(ih_v, ih_g, ihsc, FEA_);
    rownorm_kernel<<<H_, 256, 0, stream>>>(ic_v, ic_g, icsc, FEA_);
    rownorm_kernel<<<V_, 256, 0, stream>>>(fc_v, fc_g, fcsc, H_);

    tcast_kernel<<<(FEA_ * H_) / 256, 256, 0, stream>>>(ih_v, ihsc, WihT, H_, FEA_, H_);
    tcast_kernel<<<(FEA_ * H_) / 256, 256, 0, stream>>>(ic_v, icsc, WicT, H_, FEA_, H_);
    tcast_kernel<<<(H_ * VP_) / 256, 256, 0, stream>>>(fc_v, fcsc, WfcT, V_, H_, VP_);
    concat_cast_kernel<<<(KX_ * G4_) / 256, 256, 0, stream>>>(W_ih, W_hh, WgT);
    bias_sum_kernel<<<G4_ / 256, 256, 0, stream>>>(b_ih, b_hh, bsum);

    // h0 = fea @ wn(ih).T + ih_b ; c0 = fea @ wn(ic).T + ic_b
    gemm_bf16_wmma<<<dim3(H_ / 128, B_ / 128), 256, 0, stream>>>(
        fea_bf, FEA_, WihT, H_, ih_b, hbuf, H_, FEA_, H_);
    gemm_bf16_wmma<<<dim3(H_ / 128, B_ / 128), 256, 0, stream>>>(
        fea_bf, FEA_, WicT, H_, ic_b, cbuf, H_, FEA_, H_);
    cast_h_kernel<<<(B_ * H_) / 256, 256, 0, stream>>>(hbuf, xh);

    // ---- recurrence ----
    for (int t = 0; t < STEPS_; ++t) {
        pack_x_kernel<<<(B_ * E_) / 256, 256, 0, stream>>>(caption, embed_w, xh, t);
        gemm_bf16_wmma<<<dim3(G4_ / 128, B_ / 128), 256, 0, stream>>>(
            xh, KX_, WgT, G4_, bsum, gates, G4_, KX_, G4_);
        lstm_cell_kernel<<<(B_ * H_) / 256, 256, 0, stream>>>(gates, hbuf, cbuf, xh);
        gemm_bf16_wmma<<<dim3(VP_ / 128, B_ / 128), 256, 0, stream>>>(
            xh + E_, KX_, WfcT, VP_, fc_b, out + (size_t)t * V_, STEPS_ * V_, H_, V_);
    }
}
